// ModelS_87505663688867
// MI455X (gfx1250) — compile-verified, matrix operand order checked by reading, not run on hardware
//
#include <hip/hip_runtime.h>
#include <hip/hip_bf16.h>

// ---------------------------------------------------------------------------
// Types
// ---------------------------------------------------------------------------
typedef __attribute__((ext_vector_type(16))) _Float16 v16h;
typedef __attribute__((ext_vector_type(8)))  _Float16 v8h;
typedef __attribute__((ext_vector_type(8)))  float    v8f;
typedef __attribute__((ext_vector_type(4)))  float    v4f;

#define N_NODES   1024
#define M_EDGES   523776      // 1024*1023/2
#define NUM_TILES 32736       // M_EDGES / 16
#define NUM_PAIRS 16368       // NUM_TILES / 2
#define EWAVES    6           // waves per block in edge kernel (192 threads)

// wfrag offsets (in halves): fragment-formatted f16 weights
#define PE_W1 0
#define PE_W2 4096
#define PE_W3 8192
#define DE_W1 9216
#define DE_W2 13312
#define DE_W3 17408
#define WFRAG_HALVES 18432
// bias pack offsets (floats)
#define PE_B1 0
#define PE_B2 64
#define PE_B3 128
#define DE_B1 144
#define DE_B2 208
#define DE_B3 272
#define BPACK_FLOATS 288

__constant__ int IDX10c[16] = {0,1,2,3, 1,4,5,6, 2,5,7,8, 3,6,8,9};

// ---------------------------------------------------------------------------
// WMMA helpers.
// A-matrix (16x32 f16) lane layout per CDNA5 ISA 7.12.2:
//   lanes 0-15  : row M=lane,    halves hold K = {0..7, 16..23}
//   lanes 16-31 : row M=lane-16, halves hold K = {8..15, 24..31}
// B-matrix (32x16 f16) assumed mirrored with N in place of M (col per lane).
// C/D (16x16 f32): VGPR r holds M=r (lanes 0-15, N=lane) / M=r+8 (lanes 16-31).
// ---------------------------------------------------------------------------
static __device__ __forceinline__ v16h pack16(v8h lo, v8h hi) {
  v16h r;
#pragma unroll
  for (int i = 0; i < 8; i++) { r[i] = lo[i]; r[i + 8] = hi[i]; }
  return r;
}

static __device__ __forceinline__ v8f wmma_f16(v16h a, v16h b, v8f c) {
  return __builtin_amdgcn_wmma_f32_16x16x32_f16(
      /*neg_a=*/false, a, /*neg_b=*/false, b,
      /*c_mod=*/(short)0, c, /*reuse_a=*/false, /*reuse_b=*/false);
}

// One dense layer applied to TWO independent 16-row tiles sharing the same
// weights: D(16 x 16*NT) = relu( A(16x64) * W(64 x 16*NT) + b ).
// Each B fragment is loaded from LDS once and feeds two WMMAs; the two
// accumulator chains are independent, letting the scheduler fill the
// WMMA->VALU hazard slots (ISA 7.12.1) with real work instead of v_nop.
template <int NT>
static __device__ __forceinline__ void mlp_layer2(
    const v16h Aa[2], const v16h Ab[2],
    const _Float16* __restrict__ sWm, const float* __restrict__ sBm,
    int lane, v8f acca[NT], v8f accb[NT]) {
  const int col = lane & 15;
#pragma unroll
  for (int nt = 0; nt < NT; ++nt) {
    const float bv = sBm[nt * 16 + col];
    v8f ca = {bv, bv, bv, bv, bv, bv, bv, bv};
    v8f cb = ca;
#pragma unroll
    for (int kt = 0; kt < 2; ++kt) {
      const v16h bf = *(const v16h*)(sWm + ((nt * 2 + kt) * 32 + lane) * 16);
      ca = wmma_f16(Aa[kt], bf, ca);
      cb = wmma_f16(Ab[kt], bf, cb);
    }
#pragma unroll
    for (int i = 0; i < 8; i++) {
      ca[i] = fmaxf(ca[i], 0.f);
      cb[i] = fmaxf(cb[i], 0.f);
    }
    acca[nt] = ca;
    accb[nt] = cb;
  }
}

// D-tiles (2 x 4x 16x16 f32) -> f16 staging tiles [16][64] in LDS -> A frags.
// Wave-internal only: DS pipeline is in-order per wave; single dscnt==0 wait
// covers both tiles' stores before either tile's reads.
static __device__ __forceinline__ void restage2(
    const v8f ha[4], const v8f hb[4],
    _Float16* __restrict__ sta, _Float16* __restrict__ stb,
    int lane, v16h Aa[2], v16h Ab[2]) {
  const int col = lane & 15;
  const int mb  = (lane >> 4) * 8;   // D rows this half-wave owns
#pragma unroll
  for (int nt = 0; nt < 4; ++nt)
#pragma unroll
    for (int r = 0; r < 8; r++) {
      sta[(mb + r) * 64 + nt * 16 + col] = (_Float16)ha[nt][r];
      stb[(mb + r) * 64 + nt * 16 + col] = (_Float16)hb[nt][r];
    }
  asm volatile("s_wait_dscnt 0x0" ::: "memory");
  const int b0 = (lane >> 4) * 8;    // first K chunk for A layout
  const v8h* sa = (const v8h*)(sta + col * 64);
  const v8h* sb = (const v8h*)(stb + col * 64);
#pragma unroll
  for (int kt = 0; kt < 2; ++kt) {
    Aa[kt] = pack16(sa[(kt * 32 + b0) >> 3], sa[(kt * 32 + b0 + 16) >> 3]);
    Ab[kt] = pack16(sb[(kt * 32 + b0) >> 3], sb[(kt * 32 + b0 + 16) >> 3]);
  }
  asm volatile("" ::: "memory");
}

// ---------------------------------------------------------------------------
// Kernel 1: 3x SAGEConv on a complete graph (mean agg = (colsum - x)/1023)
// Single block, 1024 threads, thread t owns node t.
// ---------------------------------------------------------------------------
__global__ __launch_bounds__(1024) void conv_kernel(
    const float* __restrict__ x,
    const float* __restrict__ Wl0, const float* __restrict__ Wr0, const float* __restrict__ cb0,
    const float* __restrict__ Wl1, const float* __restrict__ Wr1, const float* __restrict__ cb1,
    const float* __restrict__ Wl2, const float* __restrict__ Wr2, const float* __restrict__ cb2,
    float* __restrict__ x3, _Float16* __restrict__ x3h, float* __restrict__ out_x) {
  __shared__ float cs[64];
  const int t = threadIdx.x;
  const int lane = t & 31;
  const float inv = 1.0f / 1023.0f;

  float a0[6];
#pragma unroll
  for (int k = 0; k < 6; k++) a0[k] = x[t * 6 + k];

  // ---- layer 0: 6 -> 32
  if (t < 64) cs[t] = 0.f;
  __syncthreads();
  for (int k = 0; k < 6; k++) {
    float v = a0[k];
    for (int m = 16; m >= 1; m >>= 1) v += __shfl_xor(v, m, 32);
    if (lane == 0) atomicAdd(&cs[k], v);
  }
  __syncthreads();
  float h1[32];
  for (int o = 0; o < 32; o++) {
    float acc = cb0[o];
    for (int k = 0; k < 6; k++) {
      float mean = (cs[k] - a0[k]) * inv;
      acc += mean * Wl0[k * 32 + o] + a0[k] * Wr0[k * 32 + o];
    }
    h1[o] = fmaxf(acc, 0.f);
  }
  __syncthreads();

  // ---- layer 1: 32 -> 32
  if (t < 64) cs[t] = 0.f;
  __syncthreads();
  for (int k = 0; k < 32; k++) {
    float v = h1[k];
    for (int m = 16; m >= 1; m >>= 1) v += __shfl_xor(v, m, 32);
    if (lane == 0) atomicAdd(&cs[k], v);
  }
  __syncthreads();
  float h2[32];
  for (int o = 0; o < 32; o++) {
    float acc = cb1[o];
    for (int k = 0; k < 32; k++) {
      float mean = (cs[k] - h1[k]) * inv;
      acc += mean * Wl1[k * 32 + o] + h1[k] * Wr1[k * 32 + o];
    }
    h2[o] = fmaxf(acc, 0.f);
  }
  __syncthreads();

  // ---- layer 2: 32 -> 64
  if (t < 64) cs[t] = 0.f;
  __syncthreads();
  for (int k = 0; k < 32; k++) {
    float v = h2[k];
    for (int m = 16; m >= 1; m >>= 1) v += __shfl_xor(v, m, 32);
    if (lane == 0) atomicAdd(&cs[k], v);
  }
  __syncthreads();
  for (int o = 0; o < 64; o++) {
    float acc = cb2[o];
    for (int k = 0; k < 32; k++) {
      float mean = (cs[k] - h2[k]) * inv;
      acc += mean * Wl2[k * 64 + o] + h2[k] * Wr2[k * 64 + o];
    }
    acc = fmaxf(acc, 0.f);
    x3[t * 64 + o]    = acc;
    out_x[t * 64 + o] = acc;           // first output of the tuple
    x3h[t * 64 + o]   = (_Float16)acc; // f16 copy for WMMA edge path
  }
}

// ---------------------------------------------------------------------------
// Kernel 2: convert a [64, Nout] fp32 weight into B-fragment f16 layout.
// Element index -> (nt, kt, lane, h); padded cols -> 0.
// ---------------------------------------------------------------------------
__global__ void prep_w_kernel(const float* __restrict__ W, int Nout, int nT,
                              _Float16* __restrict__ dst) {
  int idx = blockIdx.x * blockDim.x + threadIdx.x;
  int total = nT * 2 * 512;
  if (idx >= total) return;
  int h    = idx & 15;
  int lane = (idx >> 4) & 31;
  int kt   = (idx >> 9) & 1;
  int nt   = idx >> 10;
  int n = nt * 16 + (lane & 15);
  int klocal = (lane < 16) ? ((h < 8) ? h : h + 8)
                           : ((h < 8) ? h + 8 : h + 16);
  int k = kt * 32 + klocal;
  float v = (n < Nout) ? W[k * Nout + n] : 0.f;
  dst[idx] = (_Float16)v;
}

__global__ void prep_b_kernel(const float* __restrict__ pb1, const float* __restrict__ pb2,
                              const float* __restrict__ pb3, const float* __restrict__ db1,
                              const float* __restrict__ db2, const float* __restrict__ db3,
                              float* __restrict__ out) {
  int i = threadIdx.x;
  if (i >= BPACK_FLOATS) return;
  float v = 0.f;
  if      (i < 64)  v = pb1[i];
  else if (i < 128) v = pb2[i - 64];
  else if (i < 144) { int c = i - 128; v = (c < 10) ? pb3[c] : 0.f; }
  else if (i < 208) v = db1[i - 144];
  else if (i < 272) v = db2[i - 208];
  else              v = db3[i - 272];
  out[i] = v;
}

// ---------------------------------------------------------------------------
// Kernel 3: node MLPs pn/dn (tiny), scalar fp32, one thread per node.
// ---------------------------------------------------------------------------
static __device__ void mlp_node(const float xl[64],
    const float* __restrict__ W1, const float* __restrict__ b1,
    const float* __restrict__ W2, const float* __restrict__ b2,
    const float* __restrict__ W3, const float* __restrict__ b3,
    float* __restrict__ out10) {
  float h1[64];
  for (int o = 0; o < 64; o++) {
    float a = b1[o];
    for (int k = 0; k < 64; k++) a += xl[k] * W1[k * 64 + o];
    h1[o] = fmaxf(a, 0.f);
  }
  float h2[64];
  for (int o = 0; o < 64; o++) {
    float a = b2[o];
    for (int k = 0; k < 64; k++) a += h1[k] * W2[k * 64 + o];
    h2[o] = fmaxf(a, 0.f);
  }
  for (int c = 0; c < 10; c++) {
    float a = b3[c];
    for (int k = 0; k < 64; k++) a += h2[k] * W3[k * 10 + c];
    out10[c] = fmaxf(a, 0.f);
  }
}

__global__ __launch_bounds__(256) void node_mlp_kernel(
    const float* __restrict__ x3,
    const float* pnW1, const float* pnb1, const float* pnW2, const float* pnb2,
    const float* pnW3, const float* pnb3,
    const float* dnW1, const float* dnb1, const float* dnW2, const float* dnb2,
    const float* dnW3, const float* dnb3,
    float* __restrict__ vp, float* __restrict__ vd) {
  int t = blockIdx.x * blockDim.x + threadIdx.x;
  if (t >= N_NODES) return;
  float xl[64];
  for (int k = 0; k < 64; k++) xl[k] = x3[t * 64 + k];
  mlp_node(xl, pnW1, pnb1, pnW2, pnb2, pnW3, pnb3, vp + t * 10);
  mlp_node(xl, dnW1, dnb1, dnW2, dnb2, dnW3, dnb3, vd + t * 10);
}

// ---------------------------------------------------------------------------
// Kernel 4: fused edge MLPs (pe & de) with f16 WMMA.
// One wave processes TWO 16-edge tiles per iteration (pair), sharing each
// B fragment load across both tiles and interleaving two independent WMMA
// chains to cover the WMMA->VALU hazard slots.
// ---------------------------------------------------------------------------
__global__ __launch_bounds__(EWAVES * 32) void edge_mlp_kernel(
    const _Float16* __restrict__ x3h, const int* __restrict__ ud,
    const _Float16* __restrict__ wfrag, const float* __restrict__ bpack,
    float* __restrict__ ep, float* __restrict__ ed, int numPairs) {
  __shared__ __align__(32) _Float16 sW[WFRAG_HALVES];
  __shared__ float sB[BPACK_FLOATS];
  __shared__ __align__(32) _Float16 stage[EWAVES][2][16 * 64];

  for (int idx = threadIdx.x; idx < WFRAG_HALVES / 8; idx += blockDim.x)
    ((v8h*)sW)[idx] = ((const v8h*)wfrag)[idx];
  for (int idx = threadIdx.x; idx < BPACK_FLOATS; idx += blockDim.x)
    sB[idx] = bpack[idx];
  __syncthreads();

  const int lane = threadIdx.x & 31;
  const int wid  = threadIdx.x >> 5;
  const int col  = lane & 15;
  const int b0   = (lane >> 4) * 8;  // A-layout K-chunk base (and D row base)
  _Float16* st0 = stage[wid][0];
  _Float16* st1 = stage[wid][1];

  for (int p = blockIdx.x * EWAVES + wid; p < numPairs;
       p += gridDim.x * EWAVES) {
    const int t0 = 2 * p;
    const int t1 = 2 * p + 1;
    const int e0a = t0 * 16;
    const int e0b = t1 * 16;
    const int eia = ud[2 * (e0a + col)];
    const int eja = ud[2 * (e0a + col) + 1];
    const int eib = ud[2 * (e0b + col)];
    const int ejb = ud[2 * (e0b + col) + 1];

    // A0 = xe tiles (16x64): xe[r] = x3h[i] + x3h[j]
    v16h A0a[2], A0b[2];
    {
      const v8h* xia = (const v8h*)(x3h + eia * 64);
      const v8h* xja = (const v8h*)(x3h + eja * 64);
      const v8h* xib = (const v8h*)(x3h + eib * 64);
      const v8h* xjb = (const v8h*)(x3h + ejb * 64);
#pragma unroll
      for (int kt = 0; kt < 2; ++kt) {
        int c0 = (kt * 32 + b0) >> 3;
        A0a[kt] = pack16(xia[c0] + xja[c0], xia[c0 + 2] + xja[c0 + 2]);
        A0b[kt] = pack16(xib[c0] + xjb[c0], xib[c0 + 2] + xjb[c0 + 2]);
      }
    }

    v16h Aa[2], Ab[2];
    v8f ha[4], hb[4];
    v8f oa[1], ob[1];

    // ---- pe: 64 -> 64 -> 64 -> 10 (relu each layer)
    mlp_layer2<4>(A0a, A0b, sW + PE_W1, sB + PE_B1, lane, ha, hb);
    restage2(ha, hb, st0, st1, lane, Aa, Ab);
    mlp_layer2<4>(Aa, Ab, sW + PE_W2, sB + PE_B2, lane, ha, hb);
    restage2(ha, hb, st0, st1, lane, Aa, Ab);
    mlp_layer2<1>(Aa, Ab, sW + PE_W3, sB + PE_B3, lane, oa, ob);
    if (col < 10) {
#pragma unroll
      for (int r = 0; r < 8; r++) {
        ep[(size_t)(e0a + b0 + r) * 10 + col] = oa[0][r];
        ep[(size_t)(e0b + b0 + r) * 10 + col] = ob[0][r];
      }
    }

    // ---- de: 64 -> 64 -> 64 -> 16
    mlp_layer2<4>(A0a, A0b, sW + DE_W1, sB + DE_B1, lane, ha, hb);
    restage2(ha, hb, st0, st1, lane, Aa, Ab);
    mlp_layer2<4>(Aa, Ab, sW + DE_W2, sB + DE_B2, lane, ha, hb);
    restage2(ha, hb, st0, st1, lane, Aa, Ab);
    mlp_layer2<1>(Aa, Ab, sW + DE_W3, sB + DE_B3, lane, oa, ob);
#pragma unroll
    for (int r = 0; r < 8; r++) {
      ed[(size_t)(e0a + b0 + r) * 16 + col] = oa[0][r];
      ed[(size_t)(e0b + b0 + r) * 16 + col] = ob[0][r];
    }
  }
}

// ---------------------------------------------------------------------------
// Kernel 5: assemble X and S (coalesced float4 stores, rows of 4096).
// thread -> (row, j): writes X[row][4j..4j+3], S[row][4j..4j+3]
// ---------------------------------------------------------------------------
__global__ __launch_bounds__(256) void assemble_kernel(
    const float* __restrict__ vp, const float* __restrict__ vd,
    const float* __restrict__ ep, const float* __restrict__ ed,
    const int* __restrict__ em, float* __restrict__ X, float* __restrict__ S) {
  int tid = blockIdx.x * blockDim.x + threadIdx.x;  // 4096*1024 threads
  int j   = tid & 1023;
  int row = tid >> 10;
  int i = row >> 2, a = row & 3;
  v4f xv, sv;
  if (i == j) {
    const float* p = vp + (size_t)i * 10;
    const float* d = vd + (size_t)i * 10;
#pragma unroll
    for (int b = 0; b < 4; b++) {
      int idx = IDX10c[a * 4 + b];
      xv[b] = p[idx];
      sv[b] = d[idx];
    }
  } else {
    int eid = em[i * 1024 + j];
    const float* p = ep + (size_t)eid * 10;
    const float* d = ed + (size_t)eid * 16;
#pragma unroll
    for (int b = 0; b < 4; b++) xv[b] = p[IDX10c[a * 4 + b]];
    if (i < j) {
#pragma unroll
      for (int b = 0; b < 4; b++) sv[b] = d[a * 4 + b];
    } else {
#pragma unroll
      for (int b = 0; b < 4; b++) sv[b] = d[b * 4 + a];
    }
  }
  size_t off = (size_t)row * 4096 + (size_t)j * 4;
  *(v4f*)(X + off) = xv;
  *(v4f*)(S + off) = sv;
}

// ---------------------------------------------------------------------------
// Launch
// ---------------------------------------------------------------------------
extern "C" void kernel_launch(void* const* d_in, const int* in_sizes, int n_in,
                              void* d_out, int out_size, void* d_ws, size_t ws_size,
                              hipStream_t stream) {
  (void)in_sizes; (void)n_in; (void)out_size; (void)ws_size;
  // setup_inputs() insertion order (JAX x64 off => index arrays are int32):
  const float* x  = (const float*)d_in[0];
  // d_in[1] = edge_index (unused; complete graph handled analytically)
  const int* ud   = (const int*)d_in[2];
  const int* em   = (const int*)d_in[3];
  const float *Wl0 = (const float*)d_in[4],  *Wr0 = (const float*)d_in[5],  *cb0 = (const float*)d_in[6];
  const float *Wl1 = (const float*)d_in[7],  *Wr1 = (const float*)d_in[8],  *cb1 = (const float*)d_in[9];
  const float *Wl2 = (const float*)d_in[10], *Wr2 = (const float*)d_in[11], *cb2 = (const float*)d_in[12];
  const float *pnW1 = (const float*)d_in[13], *pnb1 = (const float*)d_in[14];
  const float *pnW2 = (const float*)d_in[15], *pnb2 = (const float*)d_in[16];
  const float *pnW3 = (const float*)d_in[17], *pnb3 = (const float*)d_in[18];
  const float *dnW1 = (const float*)d_in[19], *dnb1 = (const float*)d_in[20];
  const float *dnW2 = (const float*)d_in[21], *dnb2 = (const float*)d_in[22];
  const float *dnW3 = (const float*)d_in[23], *dnb3 = (const float*)d_in[24];
  const float *peW1 = (const float*)d_in[25], *peb1 = (const float*)d_in[26];
  const float *peW2 = (const float*)d_in[27], *peb2 = (const float*)d_in[28];
  const float *peW3 = (const float*)d_in[29], *peb3 = (const float*)d_in[30];
  const float *deW1 = (const float*)d_in[31], *deb1 = (const float*)d_in[32];
  const float *deW2 = (const float*)d_in[33], *deb2 = (const float*)d_in[34];
  const float *deW3 = (const float*)d_in[35], *deb3 = (const float*)d_in[36];

  char* ws = (char*)d_ws;
  float*    x3    = (float*)(ws + 0);                 // 256 KB
  _Float16* x3h   = (_Float16*)(ws + 262144);         // 128 KB
  float*    vp    = (float*)(ws + 393216);            // 40 KB
  float*    vd    = (float*)(ws + 434176);            // 40 KB
  _Float16* wfrag = (_Float16*)(ws + 475136);         // 36 KB
  float*    bpack = (float*)(ws + 512000);            // 1.2 KB
  float*    ep_ws = (float*)(ws + 524288);            // 20.95 MB
  float*    ed_ws = (float*)(ws + 524288 + 20951040); // 33.52 MB

  float* out_x = (float*)d_out;
  float* X = out_x + 65536;
  float* S = X + 16777216;

  conv_kernel<<<1, 1024, 0, stream>>>(x, Wl0, Wr0, cb0, Wl1, Wr1, cb1,
                                      Wl2, Wr2, cb2, x3, x3h, out_x);

  prep_w_kernel<<<16, 256, 0, stream>>>(peW1, 64, 4, wfrag + PE_W1);
  prep_w_kernel<<<16, 256, 0, stream>>>(peW2, 64, 4, wfrag + PE_W2);
  prep_w_kernel<<<4,  256, 0, stream>>>(peW3, 10, 1, wfrag + PE_W3);
  prep_w_kernel<<<16, 256, 0, stream>>>(deW1, 64, 4, wfrag + DE_W1);
  prep_w_kernel<<<16, 256, 0, stream>>>(deW2, 64, 4, wfrag + DE_W2);
  prep_w_kernel<<<4,  256, 0, stream>>>(deW3, 16, 1, wfrag + DE_W3);
  prep_b_kernel<<<1, 288, 0, stream>>>(peb1, peb2, peb3, deb1, deb2, deb3, bpack);

  node_mlp_kernel<<<4, 256, 0, stream>>>(x3, pnW1, pnb1, pnW2, pnb2, pnW3, pnb3,
                                         dnW1, dnb1, dnW2, dnb2, dnW3, dnb3,
                                         vp, vd);

  edge_mlp_kernel<<<1024, EWAVES * 32, 0, stream>>>(x3h, ud, wfrag, bpack,
                                                    ep_ws, ed_ws, NUM_PAIRS);

  assemble_kernel<<<16384, 256, 0, stream>>>(vp, vd, ep_ws, ed_ws, em, X, S);
}
